// RNN_31370441129994
// MI455X (gfx1250) — matrix operand order repaired and checked by reading, not compiled, not run on hardware
//
#include <hip/hip_runtime.h>
#include <hip/hip_bf16.h>
#include <stdint.h>

typedef __attribute__((ext_vector_type(16))) __bf16 v16bf;
typedef __attribute__((ext_vector_type(8)))  __bf16 v8bf;
typedef __attribute__((ext_vector_type(8)))  float  v8f;
typedef __attribute__((ext_vector_type(4)))  unsigned v4u;
typedef __attribute__((ext_vector_type(8)))  int      v8i;
typedef __attribute__((ext_vector_type(4)))  int      v4i;

#define MSEQ 16384
#define TT   20
#define HSZ  128
#define ESZ  64

static __device__ __forceinline__ float bf2f(__bf16 b) {
  union { unsigned short u; __bf16 b; } i; i.b = b;
  union { unsigned u; float f; } o; o.u = ((unsigned)i.u) << 16;
  return o.f;
}
static __device__ __forceinline__ __bf16 f2bf(float f) {
  union { float f; unsigned u; } a; a.f = f;
  unsigned r = a.u + 0x7FFFu + ((a.u >> 16) & 1u);
  union { unsigned short u; __bf16 b; } o; o.u = (unsigned short)(r >> 16);
  return o.b;
}
static __device__ __forceinline__ float sigm(float x) { return 1.f / (1.f + __expf(-x)); }
static __device__ __forceinline__ float tanhr(float x) { return 2.f / (1.f + __expf(-2.f * x)) - 1.f; }
static __device__ __forceinline__ v8f v8f_zero() {
  v8f v;
#pragma unroll
  for (int i = 0; i < 8; ++i) v[i] = 0.f;
  return v;
}

// Issue a TDM load of a 64 x Kin bf16 tile (global rows strided by rowStrideElems)
// into LDS at byte offset lds_off (contiguous rows). 2-D tile: groups 2/3 zero.
// This toolchain exposes the 6-arg builtin: (v4u, v8i, v4i, v4i, v8i, i32 cpol).
static __device__ __forceinline__ void tdm_load_x(const __bf16* gptr, unsigned lds_off,
                                                  int Kin, int rowStrideElems) {
  unsigned long long ga = (unsigned long long)(size_t)gptr;
  v4u g0;
  g0[0] = 1u;                                            // count=1, user mode
  g0[1] = lds_off;                                       // lds_addr
  g0[2] = (unsigned)(ga & 0xFFFFFFFFull);                // global_addr[31:0]
  g0[3] = (unsigned)((ga >> 32) & 0x01FFFFFFull)         // global_addr[56:32]
          | 0x80000000u;                                 // type=2 ("image")
  v8i g1;
  g1[0] = 0x00010000;                                    // data_size=1 (2 bytes)
  g1[1] = (int)((unsigned)Kin << 16);                    // tensor_dim0[15:0]
  g1[2] = (64 << 16);                                    // tensor_dim0[31:16]=0 | tensor_dim1 lo=64
  g1[3] = (int)((unsigned)Kin << 16);                    // tensor_dim1 hi=0 | tile_dim0=Kin
  g1[4] = 64;                                            // tile_dim1=64, tile_dim2=0
  g1[5] = rowStrideElems;                                // tensor_dim0_stride lo
  g1[6] = 0;                                             // stride hi | tensor_dim1_stride lo
  g1[7] = 0;
  v4i gz4;
  gz4[0] = 0; gz4[1] = 0; gz4[2] = 0; gz4[3] = 0;
  v8i gz8;
#pragma unroll
  for (int i = 0; i < 8; ++i) gz8[i] = 0;
  __builtin_amdgcn_tensor_load_to_lds(g0, g1, gz4, gz4, gz8, 0);
}

// ---------------- Embedding: xe = relu((x+pe) @ emb_W^T + emb_b) -> bf16 -------------
__global__ __launch_bounds__(256) void embed_kernel(
    const float* __restrict__ x, const float* __restrict__ pa,
    const float* __restrict__ pt, const float* __restrict__ eW,
    const float* __restrict__ eb, __bf16* __restrict__ xe)
{
  int idx = blockIdx.x * 256 + threadIdx.x;
  if (idx >= MSEQ * TT * ESZ) return;
  int e = idx & 63;
  int t = (idx >> 6) % TT;
  int s = idx / (64 * TT);
  int n = s & 7;
  float pe = pt[t] + pa[n];
  const float* xp = x + ((size_t)s * TT + t) * 2;
  float v = eW[e * 2 + 0] * (xp[0] + pe) + eW[e * 2 + 1] * (xp[1] + pe) + eb[e];
  xe[idx] = f2bf(v > 0.f ? v : 0.f);
}

// ------------- Repack GRU weights into extended (Kc x 512) B-fragment tiles ----------
// Wext cols: [0,256)=Wih_rz;Whh_rz fused  [256,384)=Wih_n only  [384,512)=Whh_n only
__global__ __launch_bounds__(256) void repack_gru_kernel(
    const float* __restrict__ Wih, const float* __restrict__ Whh,
    __bf16* __restrict__ pw, int Kin)
{
  int Kc = Kin + HSZ;
  int total = (Kc / 32) * 32 * 512;
  int idx = blockIdx.x * 256 + threadIdx.x;
  if (idx >= total) return;
  int e = idx & 511;
  int tile = idx >> 9;
  int ktile = tile >> 5;
  int ctile = tile & 31;
  int laneq = e >> 4;
  int i = e & 15;
  int k = ktile * 32 + ((i >> 3) << 4) + ((laneq >> 4) << 3) + (i & 7);
  int c = ctile * 16 + (laneq & 15);
  float v = 0.f;
  if (c < 256) {
    v = (k < Kin) ? Wih[(size_t)c * Kin + k] : Whh[(size_t)c * HSZ + (k - Kin)];
  } else if (c < 384) {
    if (k < Kin) v = Wih[(size_t)c * Kin + k];
  } else {
    if (k >= Kin) v = Whh[(size_t)(c - 128) * HSZ + (k - Kin)];
  }
  pw[idx] = f2bf(v);
}

__global__ __launch_bounds__(256) void repack_out_kernel(
    const float* __restrict__ oW, __bf16* __restrict__ pw)
{
  int idx = blockIdx.x * 256 + threadIdx.x;   // 4 ktiles * 4 ctiles * 512
  if (idx >= 8192) return;
  int e = idx & 511;
  int tile = idx >> 9;
  int ktile = tile >> 2;
  int ctile = tile & 3;
  int laneq = e >> 4;
  int i = e & 15;
  int k = ktile * 32 + ((i >> 3) << 4) + ((laneq >> 4) << 3) + (i & 7);
  int c = ctile * 16 + (laneq & 15);
  pw[idx] = f2bf(oW[(size_t)c * HSZ + k]);
}

__global__ void bias_pack_kernel(const float* __restrict__ bih,
                                 const float* __restrict__ bhh,
                                 float* __restrict__ bias)
{
  int idx = blockIdx.x * 256 + threadIdx.x;
  if (idx >= 3 * 512) return;
  int l = idx >> 9;
  int c = idx & 511;
  float v;
  if (c < 256)       v = bih[l * 384 + c] + bhh[l * 384 + c];
  else if (c < 384)  v = bih[l * 384 + c];
  else               v = bhh[l * 384 + (c - 128)];
  bias[idx] = v;
}

// ------------------ Fused persistent GRU layer (64 sequences / block) ----------------
// LDS: Xbuf[2][64][Kin] (TDM double buffer) | Hbuf[64][128]
__global__ __launch_bounds__(256) void gru_layer_kernel(
    const __bf16* __restrict__ xin,   // [MSEQ][TT][Kin]
    const __bf16* __restrict__ pw,    // packed Wext tiles
    const float*  __restrict__ bias,  // [512]
    __bf16* __restrict__ ys,          // [MSEQ][TT][128]
    float*  __restrict__ hfin,        // [2048][128] (agent 7 final hidden)
    int Kin)
{
  extern __shared__ __bf16 smem[];
  const int Kc = Kin + HSZ;
  const int nk = Kc >> 5;
  const int xbufElems = 64 * Kin;               // one x buffer
  __bf16* xb = smem;                            // [2][64][Kin]
  __bf16* hb = smem + 2 * xbufElems;            // [64][128]
  const int tid = threadIdx.x;
  const int lane = tid & 31;
  const int w = tid >> 5;
  const int rt = w >> 1;
  const int gsel = w & 1;
  const int s0 = blockIdx.x * 64;

  __builtin_prefetch(pw, 0, 3);                 // pull packed weights toward caches

  // kick off TDM load of x_0 into Xbuf[0]
  if (w == 0) {
    tdm_load_x(xin + ((size_t)s0 * TT + 0) * Kin, 0u, Kin, TT * Kin);
  }
  // zero hidden state while the DMA runs (disjoint LDS region)
  for (int idx = tid; idx < 64 * 64; idx += 256) {
    int r = idx >> 6, cu = idx & 63;
    reinterpret_cast<unsigned*>(hb + r * HSZ)[cu] = 0u;
  }

  const int arow = rt * 16 + (lane & 15);
  const int ksel8 = (lane >> 4) << 3;
  const int prow = rt * 16 + (lane >> 4) * 8;
  const int colL = lane & 15;
  float hkeep[4][8];

  for (int t = 0; t < TT; ++t) {
    if (w == 0) __builtin_amdgcn_s_wait_tensorcnt(0);   // x_t tile landed in LDS
    __syncthreads();                                    // publish x_t (and h writes)
    if (w == 0 && t + 1 < TT) {                          // prefetch x_{t+1} into other buffer
      tdm_load_x(xin + ((size_t)s0 * TT + (t + 1)) * Kin,
                 (unsigned)(((t + 1) & 1) * xbufElems * 2), Kin, TT * Kin);
    }
    const __bf16* xcur = xb + (t & 1) * xbufElems;

    v8f acc[16];
#pragma unroll
    for (int i = 0; i < 16; ++i) acc[i] = v8f_zero();

    for (int ks = 0; ks < nk; ++ks) {
      int kbase = ks * 32;
      const __bf16* ab = (kbase < Kin)
          ? (xcur + arow * Kin + kbase + ksel8)
          : (hb + arow * HSZ + (kbase - Kin) + ksel8);
      union { v16bf v; v8bf h[2]; } au;
      au.h[0] = *reinterpret_cast<const v8bf*>(ab);
      au.h[1] = *reinterpret_cast<const v8bf*>(ab + 16);
      const v16bf* bp = reinterpret_cast<const v16bf*>(pw) + (size_t)ks * 32 * 32;
#pragma unroll
      for (int j = 0; j < 4; ++j) {
        int ctl = gsel * 4 + j;
        v16bf br = bp[(ctl)      * 32 + lane];
        v16bf bz = bp[(ctl + 8)  * 32 + lane];
        v16bf bi = bp[(ctl + 16) * 32 + lane];
        v16bf bh = bp[(ctl + 24) * 32 + lane];
        acc[j]      = __builtin_amdgcn_wmma_f32_16x16x32_bf16(false, au.v, false, br, (short)0, acc[j],      false, false);
        acc[4 + j]  = __builtin_amdgcn_wmma_f32_16x16x32_bf16(false, au.v, false, bz, (short)0, acc[4 + j],  false, false);
        acc[8 + j]  = __builtin_amdgcn_wmma_f32_16x16x32_bf16(false, au.v, false, bi, (short)0, acc[8 + j],  false, false);
        acc[12 + j] = __builtin_amdgcn_wmma_f32_16x16x32_bf16(false, au.v, false, bh, (short)0, acc[12 + j], false, false);
      }
    }

    // gate math (reads old h from LDS; no LDS writes before the barrier)
#pragma unroll
    for (int j = 0; j < 4; ++j) {
      int c = gsel * 64 + j * 16 + colL;        // 0..127
      float br = bias[c];
      float bz = bias[128 + c];
      float bi = bias[256 + c];
      float bh = bias[384 + c];
#pragma unroll
      for (int e = 0; e < 8; ++e) {
        int row = prow + e;
        float hprev = bf2f(hb[row * HSZ + c]);
        float r = sigm(acc[j][e] + br);
        float z = sigm(acc[4 + j][e] + bz);
        float n = tanhr(acc[8 + j][e] + bi + r * (acc[12 + j][e] + bh));
        hkeep[j][e] = (1.f - z) * n + z * hprev;
      }
    }
    __syncthreads();
#pragma unroll
    for (int j = 0; j < 4; ++j) {
      int c = gsel * 64 + j * 16 + colL;
#pragma unroll
      for (int e = 0; e < 8; ++e) {
        int row = prow + e;
        float hv = hkeep[j][e];
        hb[row * HSZ + c] = f2bf(hv);
        int s = s0 + row;
        ys[((size_t)s * TT + t) * HSZ + c] = f2bf(hv);
        if (t == TT - 1 && (s & 7) == 7)
          hfin[(size_t)(s >> 3) * HSZ + c] = hv;
      }
    }
  }
}

// ---------------------- Output GEMM: out = alpha @ out_W^T + out_b ------------------
__global__ __launch_bounds__(256) void out_gemm_kernel(
    const __bf16* __restrict__ ysL, const __bf16* __restrict__ pwo,
    const float* __restrict__ ob, float* __restrict__ out)
{
  __shared__ __align__(16) __bf16 smem[64 * HSZ];
  int tid = threadIdx.x, lane = tid & 31, w = tid >> 5;
  int rt = w >> 1, cs = w & 1;
  int s0 = blockIdx.x * 64;
  for (int idx = tid; idx < 64 * 16; idx += 256) {
    int r = idx >> 4, cc = (idx & 15) << 3;
    *reinterpret_cast<v8bf*>(smem + r * HSZ + cc) =
        *reinterpret_cast<const v8bf*>(ysL + ((size_t)(s0 + r) * TT + (TT - 1)) * HSZ + cc);
  }
  __syncthreads();
  v8f acc[2];
  acc[0] = v8f_zero(); acc[1] = v8f_zero();
  int arow = rt * 16 + (lane & 15);
  int ksel8 = (lane >> 4) << 3;
  for (int ks = 0; ks < 4; ++ks) {
    const __bf16* ab = smem + arow * HSZ + ks * 32 + ksel8;
    union { v16bf v; v8bf h[2]; } au;
    au.h[0] = *reinterpret_cast<const v8bf*>(ab);
    au.h[1] = *reinterpret_cast<const v8bf*>(ab + 16);
    const v16bf* bp = reinterpret_cast<const v16bf*>(pwo);
#pragma unroll
    for (int j = 0; j < 2; ++j) {
      int ct = cs * 2 + j;
      v16bf b = bp[((size_t)ks * 4 + ct) * 32 + lane];
      acc[j] = __builtin_amdgcn_wmma_f32_16x16x32_bf16(false, au.v, false, b, (short)0, acc[j], false, false);
    }
  }
  int prow = rt * 16 + (lane >> 4) * 8;
  int colL = lane & 15;
#pragma unroll
  for (int j = 0; j < 2; ++j) {
    int c = (cs * 2 + j) * 16 + colL;          // 0..63
    float bb = ob[c];
#pragma unroll
    for (int e = 0; e < 8; ++e) {
      int s = s0 + prow + e;
      out[(size_t)s * 64 + c] = acc[j][e] + bb;
    }
  }
}

// ------------------------------------- Launch ---------------------------------------
extern "C" void kernel_launch(void* const* d_in, const int* in_sizes, int n_in,
                              void* d_out, int out_size, void* d_ws, size_t ws_size,
                              hipStream_t stream) {
  (void)in_sizes; (void)n_in; (void)out_size; (void)ws_size;
  const float* x        = (const float*)d_in[0];
  const float* pa       = (const float*)d_in[1];
  const float* pt       = (const float*)d_in[2];
  const float* eW       = (const float*)d_in[3];
  const float* eb       = (const float*)d_in[4];
  const float* Wih0     = (const float*)d_in[5];
  const float* Wih_rest = (const float*)d_in[6];
  const float* Whh      = (const float*)d_in[7];
  const float* bih      = (const float*)d_in[8];
  const float* bhh      = (const float*)d_in[9];
  const float* out_W    = (const float*)d_in[10];
  const float* out_b    = (const float*)d_in[11];

  char* ws = (char*)d_ws;
  size_t off_xe  = 0;
  size_t off_ys0 = off_xe  + (size_t)MSEQ * TT * ESZ * 2;
  size_t off_ys1 = off_ys0 + (size_t)MSEQ * TT * HSZ * 2;
  size_t off_pw0 = off_ys1 + (size_t)MSEQ * TT * HSZ * 2;
  size_t off_pw1 = off_pw0 + (size_t)6 * 32 * 512 * 2;
  size_t off_pw2 = off_pw1 + (size_t)8 * 32 * 512 * 2;
  size_t off_pwo = off_pw2 + (size_t)8 * 32 * 512 * 2;
  size_t off_bs  = off_pwo + (size_t)16 * 512 * 2;

  __bf16* xe  = (__bf16*)(ws + off_xe);
  __bf16* ys0 = (__bf16*)(ws + off_ys0);
  __bf16* ys1 = (__bf16*)(ws + off_ys1);
  __bf16* pw0 = (__bf16*)(ws + off_pw0);
  __bf16* pw1 = (__bf16*)(ws + off_pw1);
  __bf16* pw2 = (__bf16*)(ws + off_pw2);
  __bf16* pwo = (__bf16*)(ws + off_pwo);
  float*  bsb = (float*)(ws + off_bs);

  float* out = (float*)d_out;
  float* hid = out + (size_t)2048 * 512;   // hidden section: [3][2048][128]

  embed_kernel<<<(MSEQ * TT * ESZ) / 256, 256, 0, stream>>>(x, pa, pt, eW, eb, xe);

  repack_gru_kernel<<<384, 256, 0, stream>>>(Wih0, Whh, pw0, ESZ);
  repack_gru_kernel<<<512, 256, 0, stream>>>(Wih_rest, Whh + 384 * HSZ, pw1, HSZ);
  repack_gru_kernel<<<512, 256, 0, stream>>>(Wih_rest + 384 * HSZ, Whh + 2 * 384 * HSZ, pw2, HSZ);
  repack_out_kernel<<<32, 256, 0, stream>>>(out_W, pwo);
  bias_pack_kernel<<<6, 256, 0, stream>>>(bih, bhh, bsb);

  // dynamic LDS: 2*64*Kin*2 (x double buffer) + 64*128*2 (hidden)
  gru_layer_kernel<<<MSEQ / 64, 256, 2 * 64 * ESZ * 2 + 64 * HSZ * 2, stream>>>(
      xe, pw0, bsb, ys0, hid, ESZ);
  gru_layer_kernel<<<MSEQ / 64, 256, 2 * 64 * HSZ * 2 + 64 * HSZ * 2, stream>>>(
      ys0, pw1, bsb + 512, ys1, hid + (size_t)2048 * HSZ, HSZ);
  gru_layer_kernel<<<MSEQ / 64, 256, 2 * 64 * HSZ * 2 + 64 * HSZ * 2, stream>>>(
      ys1, pw2, bsb + 1024, ys0, hid + (size_t)2 * 2048 * HSZ, HSZ);

  out_gemm_kernel<<<MSEQ / 64, 256, 0, stream>>>(ys0, pwo, out_b, out);
}
